// BiasedMHSA_12816182411331
// MI455X (gfx1250) — matrix operand order repaired and checked by reading, not compiled, not run on hardware
//
#include <hip/hip_runtime.h>

// ---------------------------------------------------------------------------
// Types matching gfx1250 WMMA operand layouts (CDNA5 ISA)
// ---------------------------------------------------------------------------
typedef __bf16 bf16_t;
typedef __attribute__((ext_vector_type(8)))  __bf16 v8bf;
typedef __attribute__((ext_vector_type(16))) __bf16 v16bf;
typedef __attribute__((ext_vector_type(8)))  float  v8f;

#define D_MODEL 512
#define N_HEADS 8
#define D_HEAD  64
#define BATCH   4
#define SEQ     2048
#define ROWS    (BATCH * SEQ)   // 8192

// ---------------------------------------------------------------------------
// Fragment loaders (bf16, 16x16x32 WMMA)
//
// A-matrix 16x32 (MxK), per lane: row m = lane&15, half h = lane>>4
//   vec[i] = A[m][ 16*h + i ]  -> two 16-byte chunks.
// B-matrix 32x16 (KxN), per lane: col n = lane&15, half h = lane>>4
//   vec[i] = B[16*h + i][n] -> one contiguous 32-byte chunk from a
//   row-major (n, k) source.
// C/D 16x16 f32: col n = lane&15, row = r + 8*(lane>>4)
// ---------------------------------------------------------------------------
__device__ static inline v16bf load_frag_a(const bf16_t* base, int ld, int lane) {
  const int row = lane & 15;
  const int h   = lane >> 4;
  const bf16_t* p0 = base + (size_t)row * ld + h * 16;
  const v8bf lo = *(const v8bf*)(p0);
  const v8bf hi = *(const v8bf*)(p0 + 8);
  v16bf a;
#pragma unroll
  for (int i = 0; i < 8; ++i) { a[i] = lo[i]; a[i + 8] = hi[i]; }
  return a;
}

// base points at element (n_row = 0, k = 0) of a row-major (n, k) tile
__device__ static inline v16bf load_frag_b(const bf16_t* base, int ld, int lane) {
  const int n = lane & 15;
  const int h = lane >> 4;
  return *(const v16bf*)(base + (size_t)n * ld + h * 16);
}

__device__ static inline v8f wmma_bf16(v16bf a, v16bf b, v8f c) {
  return __builtin_amdgcn_wmma_f32_16x16x32_bf16(
      /*neg_a=*/false, a, /*neg_b=*/false, b,
      /*c_mod=*/(short)0, c, /*reuse_a=*/false, /*reuse_b=*/false);
}

// ---------------------------------------------------------------------------
// Stage 0: precision conversion / layout kernels
// ---------------------------------------------------------------------------
__global__ void f32_to_bf16_kernel(const float* __restrict__ src,
                                   bf16_t* __restrict__ dst, int n) {
  int i = blockIdx.x * 256 + threadIdx.x;
  if (i < n) dst[i] = (bf16_t)src[i];
}

// w stored (in, out) row-major -> wT (out, in) row-major, bf16
__global__ void transpose_w_kernel(const float* __restrict__ w,
                                   bf16_t* __restrict__ wT) {
  int idx = blockIdx.x * 256 + threadIdx.x;     // over 512*512
  int i = idx >> 9;        // input dim
  int o = idx & 511;       // output dim
  wT[(size_t)o * D_MODEL + i] = (bf16_t)w[idx];
}

// ---------------------------------------------------------------------------
// Stage 1: fused QKV projection.
// Block = 256 threads (8 waves, 4x2), block tile 128x128, wave tile 32x64.
// Writes Q (scaled 1/sqrt(dh)) and K as (B,H,N,dh) bf16; V transposed as
// (B,H,dh,N) bf16 so attention B-fragments are contiguous.
// ---------------------------------------------------------------------------
__global__ void __launch_bounds__(256)
qkv_gemm_kernel(const bf16_t* __restrict__ xb,
                const bf16_t* __restrict__ wqT, const bf16_t* __restrict__ wkT,
                const bf16_t* __restrict__ wvT,
                const float* __restrict__ bq, const float* __restrict__ bk,
                const float* __restrict__ bv,
                bf16_t* __restrict__ Q, bf16_t* __restrict__ K,
                bf16_t* __restrict__ VT) {
  const int lane = threadIdx.x & 31;
  const int wave = threadIdx.x >> 5;           // 0..7
  const int wm = wave & 3, wn = wave >> 2;     // 4 x 2 wave grid
  const int m0 = blockIdx.x * 128 + wm * 32;
  const int n0 = blockIdx.y * 128 + wn * 64;
  const int z  = blockIdx.z;                   // 0:Q 1:K 2:V (uniform)

  const bf16_t* wT  = (z == 0) ? wqT : (z == 1) ? wkT : wvT;
  const float* bias = (z == 0) ? bq  : (z == 1) ? bk  : bv;

  v8f acc[2][4] = {};
#pragma unroll 1
  for (int k0 = 0; k0 < D_MODEL; k0 += 32) {
    v16bf a0 = load_frag_a(xb + (size_t)m0 * D_MODEL + k0, D_MODEL, lane);
    v16bf a1 = load_frag_a(xb + (size_t)(m0 + 16) * D_MODEL + k0, D_MODEL, lane);
#pragma unroll
    for (int nt = 0; nt < 4; ++nt) {
      v16bf b = load_frag_b(wT + (size_t)(n0 + nt * 16) * D_MODEL + k0,
                            D_MODEL, lane);
      acc[0][nt] = wmma_bf16(a0, b, acc[0][nt]);
      acc[1][nt] = wmma_bf16(a1, b, acc[1][nt]);
    }
  }

  const int cn = lane & 15, ch = lane >> 4;
#pragma unroll
  for (int mt = 0; mt < 2; ++mt) {
#pragma unroll
    for (int nt = 0; nt < 4; ++nt) {
      const int n = n0 + nt * 16 + cn;
      const float bn = bias[n];
      const int hh = n >> 6, d = n & 63;
#pragma unroll
      for (int r = 0; r < 8; ++r) {
        const int m = m0 + mt * 16 + r + 8 * ch;
        const int b_ = m >> 11;          // batch
        const int tok = m & (SEQ - 1);   // token
        const float val = acc[mt][nt][r] + bn;
        const size_t bh = (size_t)b_ * N_HEADS + hh;
        if (z == 0)
          Q[(bh * SEQ + tok) * D_HEAD + d] = (bf16_t)(val * 0.125f); // 1/sqrt(64)
        else if (z == 1)
          K[(bh * SEQ + tok) * D_HEAD + d] = (bf16_t)val;
        else
          VT[(bh * D_HEAD + d) * SEQ + tok] = (bf16_t)val;
      }
    }
  }
}

// ---------------------------------------------------------------------------
// Stage 2: flash attention with block-cooperative K/V staging.
// One block (8 waves) owns (b, h, 128 query rows); each wave owns a 16-row
// q tile. K/V stream through double-buffered LDS chunks of 64 keys so every
// chunk is fetched from L2/HBM once per 128 query rows (8x traffic cut).
// Next chunk's global_load_b128s are issued before the compute section so
// they overlap the 16 WMMAs + softmax of the current chunk.
// ---------------------------------------------------------------------------
__global__ void __launch_bounds__(256)
attn_kernel(const bf16_t* __restrict__ Q, const bf16_t* __restrict__ K,
            const bf16_t* __restrict__ VT, const float* __restrict__ bias,
            bf16_t* __restrict__ ctx) {
  __shared__ __align__(32) bf16_t kbuf[2][64 * 64];   // (key, dh) chunk
  __shared__ __align__(32) bf16_t vbuf[2][64 * 64];   // (dh, key) chunk
  __shared__ __align__(32) bf16_t pshared[8][16 * 64];

  const int tid  = threadIdx.x;
  const int lane = tid & 31;
  const int wave = tid >> 5;
  const int NQB  = SEQ / 128;                 // 16 q-blocks per (b,h)
  const int bh   = blockIdx.x / NQB;          // 0..31
  const int qb   = blockIdx.x % NQB;
  const int q0   = qb * 128 + wave * 16;
  const int b    = bh >> 3;
  const int hsel = bh & 7;

  const bf16_t* Qp = Q  + (size_t)bh * SEQ * D_HEAD;
  const bf16_t* Kp = K  + (size_t)bh * SEQ * D_HEAD;
  const bf16_t* Vp = VT + (size_t)bh * D_HEAD * SEQ;
  const float*  Bp = bias + (size_t)b * SEQ * SEQ;

  // cooperative-copy indexing: 256 threads x 32B cover one 8KB chunk
  const int kofs = tid * 16;                  // K chunk is fully contiguous
  const int vrow = tid >> 2;                  // dh row 0..63
  const int vcol = (tid & 3) * 16;            // key sub-offset

  // ---- prologue: stage chunk 0 into buffer 0 ---------------------------
  {
    v16bf kr = *(const v16bf*)(Kp + kofs);
    v16bf vr = *(const v16bf*)(Vp + (size_t)vrow * SEQ + vcol);
    *(v16bf*)(kbuf[0] + kofs) = kr;
    *(v16bf*)(vbuf[0] + vrow * 64 + vcol) = vr;
  }
  __syncthreads();

  // Q fragments for this 16-row tile (dh = 64 -> two K=32 steps)
  const v16bf qa0 = load_frag_a(Qp + (size_t)q0 * D_HEAD + 0,  D_HEAD, lane);
  const v16bf qa1 = load_frag_a(Qp + (size_t)q0 * D_HEAD + 32, D_HEAD, lane);

  v8f of[4] = {};
  float mrow[8], lrow[8];
#pragma unroll
  for (int r = 0; r < 8; ++r) { mrow[r] = -__builtin_inff(); lrow[r] = 0.0f; }

  bf16_t* pbuf = pshared[wave];
  const int cn = lane & 15, ch = lane >> 4;
  int cur = 0;

#pragma unroll 1
  for (int kt = 0; kt < SEQ; kt += 64) {
    const bool more = (kt + 64) < SEQ;
    // issue next chunk's global loads early (latency hidden by compute)
    v16bf kr, vr;
    if (more) {
      kr = *(const v16bf*)(Kp + (size_t)(kt + 64) * D_HEAD + kofs);
      vr = *(const v16bf*)(Vp + (size_t)vrow * SEQ + (kt + 64) + vcol);
    }
    const bf16_t* Kl = kbuf[cur];
    const bf16_t* Vl = vbuf[cur];

    // ---- S = Q K^T for a 16x64 tile (from LDS) ------------------------
    v8f s[4];
#pragma unroll
    for (int nt = 0; nt < 4; ++nt) {
      v8f c = {};
      c = wmma_bf16(qa0, load_frag_b(Kl + (nt * 16) * 64 + 0,  64, lane), c);
      c = wmma_bf16(qa1, load_frag_b(Kl + (nt * 16) * 64 + 32, 64, lane), c);
      s[nt] = c;
    }
    // ---- + bias (fp32, C-layout loads; per-batch slab is L2-resident) --
#pragma unroll
    for (int nt = 0; nt < 4; ++nt)
#pragma unroll
      for (int r = 0; r < 8; ++r)
        s[nt][r] += Bp[(size_t)(q0 + r + 8 * ch) * SEQ + kt + nt * 16 + cn];

    // ---- online softmax ----------------------------------------------
    float corr[8];
#pragma unroll
    for (int r = 0; r < 8; ++r) {
      float v = s[0][r];
      v = fmaxf(v, s[1][r]); v = fmaxf(v, s[2][r]); v = fmaxf(v, s[3][r]);
      v = fmaxf(v, __shfl_xor(v, 1, 32));
      v = fmaxf(v, __shfl_xor(v, 2, 32));
      v = fmaxf(v, __shfl_xor(v, 4, 32));
      v = fmaxf(v, __shfl_xor(v, 8, 32));          // row max over 64 keys
      const float mn = fmaxf(mrow[r], v);
      corr[r] = __expf(mrow[r] - mn);
      mrow[r] = mn;
    }
    float rs[8];
#pragma unroll
    for (int r = 0; r < 8; ++r) rs[r] = 0.0f;
#pragma unroll
    for (int nt = 0; nt < 4; ++nt)
#pragma unroll
      for (int r = 0; r < 8; ++r) {
        const float p = __expf(s[nt][r] - mrow[r]);
        s[nt][r] = p;
        rs[r] += p;
      }
#pragma unroll
    for (int r = 0; r < 8; ++r) {
      float v = rs[r];
      v += __shfl_xor(v, 1, 32);
      v += __shfl_xor(v, 2, 32);
      v += __shfl_xor(v, 4, 32);
      v += __shfl_xor(v, 8, 32);
      lrow[r] = lrow[r] * corr[r] + v;
    }
#pragma unroll
    for (int nt = 0; nt < 4; ++nt)
#pragma unroll
      for (int r = 0; r < 8; ++r) of[nt][r] *= corr[r];

    // ---- P: C-layout -> per-wave LDS slab -> A-layout fragments --------
#pragma unroll
    for (int nt = 0; nt < 4; ++nt)
#pragma unroll
      for (int r = 0; r < 8; ++r)
        pbuf[(r + 8 * ch) * 64 + nt * 16 + cn] = (bf16_t)s[nt][r];
    // same-wave LDS RAW: compiler inserts s_wait_dscnt
    const v16bf pa0 = load_frag_a(pbuf + 0,  64, lane);
    const v16bf pa1 = load_frag_a(pbuf + 32, 64, lane);

    // ---- O += P * V^T (from LDS) ---------------------------------------
#pragma unroll
    for (int nt = 0; nt < 4; ++nt) {
      of[nt] = wmma_bf16(pa0, load_frag_b(Vl + (nt * 16) * 64 + 0,  64, lane), of[nt]);
      of[nt] = wmma_bf16(pa1, load_frag_b(Vl + (nt * 16) * 64 + 32, 64, lane), of[nt]);
    }

    // ---- stage next chunk into the alternate buffer --------------------
    if (more) {
      *(v16bf*)(kbuf[cur ^ 1] + kofs) = kr;
      *(v16bf*)(vbuf[cur ^ 1] + vrow * 64 + vcol) = vr;
    }
    __syncthreads();   // orders this iter's reads/writes vs next iter
    cur ^= 1;
  }

  // ---- finalize: O /= l, write ctx (B, N, H*dh) bf16 -------------------
#pragma unroll
  for (int r = 0; r < 8; ++r) lrow[r] = 1.0f / lrow[r];
#pragma unroll
  for (int nt = 0; nt < 4; ++nt)
#pragma unroll
    for (int r = 0; r < 8; ++r) {
      const int tok = q0 + r + 8 * ch;
      ctx[((size_t)b * SEQ + tok) * D_MODEL + hsel * D_HEAD + nt * 16 + cn] =
          (bf16_t)(of[nt][r] * lrow[r]);
    }
}

// ---------------------------------------------------------------------------
// Stage 3: output projection, fp32 result + bias.
// ---------------------------------------------------------------------------
__global__ void __launch_bounds__(256)
out_gemm_kernel(const bf16_t* __restrict__ ctx, const bf16_t* __restrict__ woT,
                const float* __restrict__ bo, float* __restrict__ out) {
  const int lane = threadIdx.x & 31;
  const int wave = threadIdx.x >> 5;
  const int wm = wave & 3, wn = wave >> 2;
  const int m0 = blockIdx.x * 128 + wm * 32;
  const int n0 = blockIdx.y * 128 + wn * 64;

  v8f acc[2][4] = {};
#pragma unroll 1
  for (int k0 = 0; k0 < D_MODEL; k0 += 32) {
    v16bf a0 = load_frag_a(ctx + (size_t)m0 * D_MODEL + k0, D_MODEL, lane);
    v16bf a1 = load_frag_a(ctx + (size_t)(m0 + 16) * D_MODEL + k0, D_MODEL, lane);
#pragma unroll
    for (int nt = 0; nt < 4; ++nt) {
      v16bf b = load_frag_b(woT + (size_t)(n0 + nt * 16) * D_MODEL + k0,
                            D_MODEL, lane);
      acc[0][nt] = wmma_bf16(a0, b, acc[0][nt]);
      acc[1][nt] = wmma_bf16(a1, b, acc[1][nt]);
    }
  }

  const int cn = lane & 15, ch = lane >> 4;
#pragma unroll
  for (int mt = 0; mt < 2; ++mt)
#pragma unroll
    for (int nt = 0; nt < 4; ++nt) {
      const int n = n0 + nt * 16 + cn;
      const float bn = bo[n];
#pragma unroll
      for (int r = 0; r < 8; ++r) {
        const int m = m0 + mt * 16 + r + 8 * ch;
        out[(size_t)m * D_MODEL + n] = acc[mt][nt][r] + bn;
      }
    }
}

// ---------------------------------------------------------------------------
// Host launcher
// ---------------------------------------------------------------------------
extern "C" void kernel_launch(void* const* d_in, const int* in_sizes, int n_in,
                              void* d_out, int out_size, void* d_ws,
                              size_t ws_size, hipStream_t stream) {
  const float* x    = (const float*)d_in[0];
  const float* abp  = (const float*)d_in[1];
  const float* wq   = (const float*)d_in[2];
  const float* bq   = (const float*)d_in[3];
  const float* wk   = (const float*)d_in[4];
  const float* bk   = (const float*)d_in[5];
  const float* wv   = (const float*)d_in[6];
  const float* bv   = (const float*)d_in[7];
  const float* wo   = (const float*)d_in[8];
  const float* bo   = (const float*)d_in[9];
  float* out        = (float*)d_out;

  char* ws = (char*)d_ws;
  size_t off = 0;
  bf16_t* xb  = (bf16_t*)(ws + off); off += (size_t)ROWS * D_MODEL * 2;     // 8 MB
  bf16_t* wqT = (bf16_t*)(ws + off); off += (size_t)D_MODEL * D_MODEL * 2;
  bf16_t* wkT = (bf16_t*)(ws + off); off += (size_t)D_MODEL * D_MODEL * 2;
  bf16_t* wvT = (bf16_t*)(ws + off); off += (size_t)D_MODEL * D_MODEL * 2;
  bf16_t* woT = (bf16_t*)(ws + off); off += (size_t)D_MODEL * D_MODEL * 2;
  bf16_t* Qb  = (bf16_t*)(ws + off); off += (size_t)ROWS * D_MODEL * 2;     // (B,H,N,dh)
  bf16_t* Kb  = (bf16_t*)(ws + off); off += (size_t)ROWS * D_MODEL * 2;
  bf16_t* VTb = (bf16_t*)(ws + off); off += (size_t)ROWS * D_MODEL * 2;     // (B,H,dh,N)
  bf16_t* ctx = (bf16_t*)(ws + off); off += (size_t)ROWS * D_MODEL * 2;     // (B,N,D)

  // Stage 0: convert / transpose to bf16
  const int nx = ROWS * D_MODEL;
  f32_to_bf16_kernel<<<nx / 256, 256, 0, stream>>>(x, xb, nx);
  transpose_w_kernel<<<(D_MODEL * D_MODEL) / 256, 256, 0, stream>>>(wq, wqT);
  transpose_w_kernel<<<(D_MODEL * D_MODEL) / 256, 256, 0, stream>>>(wk, wkT);
  transpose_w_kernel<<<(D_MODEL * D_MODEL) / 256, 256, 0, stream>>>(wv, wvT);
  transpose_w_kernel<<<(D_MODEL * D_MODEL) / 256, 256, 0, stream>>>(wo, woT);

  // Stage 1: QKV projection (grid.z selects Q/K/V)
  qkv_gemm_kernel<<<dim3(ROWS / 128, D_MODEL / 128, 3), 256, 0, stream>>>(
      xb, wqT, wkT, wvT, bq, bk, bv, Qb, Kb, VTb);

  // Stage 2: flash attention, one block per (b, h, 128 query rows)
  attn_kernel<<<BATCH * N_HEADS * (SEQ / 128), 256, 0, stream>>>(
      Qb, Kb, VTb, abp, ctx);

  // Stage 3: output projection
  out_gemm_kernel<<<dim3(ROWS / 128, D_MODEL / 128, 1), 256, 0, stream>>>(
      ctx, woT, bo, out);
}
